// NextVisitTime_62818191671401
// MI455X (gfx1250) — matrix operand order repaired and checked by reading, not compiled, not run on hardware
//
#include <hip/hip_runtime.h>

// ---------------------------------------------------------------------------
// NextVisitTime fused attention for gfx1250 (MI455X), wave32 + WMMA f16.
//
// Algebraic restructure:
//   q[n] = QU[b] + P[hour[n]]          (Q projection folded into tables)
//   QU   = user_pref[user] @ Wq[0:256] (512x256, computed once)
//   P    = E @ Wq[256:512] + bq        (24x256)
//   k,v  = E @ Wk/Wv + b               (24x256)
//   out  = softmax(q k^T * s, mask) v  @ Wu + bu   (WMMA f16, f32 accum)
//
// Scores are computed TRANSPOSED (scoresT = K x q^T) so each lane owns one
// query row's softmax: 16 in-register ops + one ds_swizzle (lane^16).
// 32 rows per block (2 tiles/wave) so each Wu B-tile load feeds two WMMAs.
// ---------------------------------------------------------------------------

#define NH 4
#define HD 64
#define DD 256
#define TT 24
#define TP 32
#define NN 65536   // B*S = 512*128
#define KT8 8      // K tiles in heads@Wu : 256/32

typedef __attribute__((ext_vector_type(16))) _Float16 v16h;
typedef __attribute__((ext_vector_type(8)))  _Float16 v8h;
typedef __attribute__((ext_vector_type(8)))  float    v8f;

static __device__ inline v16h cat8(v8h lo, v8h hi) {
  return __builtin_shufflevector(lo, hi, 0,1,2,3,4,5,6,7,8,9,10,11,12,13,14,15);
}

// A-matrix (16x32 f16, MxK) load from row-major halves.
// Contiguous K: lanes 0-15 hold row M=lane, K = 0..15;
//             lanes 16-31 hold row M=lane-16, K = 16..31.
static __device__ inline v16h load_a_rm(const _Float16* base, int ld, int lane) {
  int row = lane & 15;
  int kb  = (lane >> 4) << 4;              // 0 or 16
  const _Float16* p = base + row * ld + kb;
  v8h lo = *(const v8h*)(p);               // K = kb .. kb+7
  v8h hi = *(const v8h*)(p + 8);           // K = kb+8 .. kb+15
  return cat8(lo, hi);
}

// Pre-packed operand tile: 32 lanes x 16 halves, contiguous per lane.
static __device__ inline v16h load_pk(const _Float16* blob, int tile, int lane) {
  const _Float16* p = blob + (((tile << 5) + lane) << 4);
  v8h lo = *(const v8h*)(p);
  v8h hi = *(const v8h*)(p + 8);
  return cat8(lo, hi);
}

// lane <-> lane^16 exchange: ds_swizzle 32-lane mode, and=0x1f, xor=0x10.
static __device__ inline float swz16(float x) {
  return __int_as_float(__builtin_amdgcn_ds_swizzle(__float_as_int(x), 0x401f));
}

// ---------------------------------------------------------------------------
// K1: k = E@Wk+bk, v = E@Wv+bv, P = E@Wq[256:512]+bq     (24 blocks x 256 thr)
// ---------------------------------------------------------------------------
__global__ void kv_p_kernel(const float* __restrict__ E,  const float* __restrict__ Wq,
                            const float* __restrict__ bq, const float* __restrict__ Wk,
                            const float* __restrict__ bk, const float* __restrict__ Wv,
                            const float* __restrict__ bv,
                            float* __restrict__ Kf, float* __restrict__ Vf,
                            float* __restrict__ P) {
  __shared__ float e[DD];
  int t = blockIdx.x, j = threadIdx.x;
  e[j] = E[t * DD + j];
  __syncthreads();
  float ak = 0.f, av = 0.f, ap = 0.f;
#pragma unroll 4
  for (int d = 0; d < DD; ++d) {
    float ed = e[d];
    ak = fmaf(ed, Wk[d * DD + j], ak);
    av = fmaf(ed, Wv[d * DD + j], av);
    ap = fmaf(ed, Wq[(DD + d) * DD + j], ap);
  }
  Kf[t * DD + j] = ak + bk[j];
  Vf[t * DD + j] = av + bv[j];
  P [t * DD + j] = ap + bq[j];
}

// ---------------------------------------------------------------------------
// K2: QU[b] = user_pref[user[b]] @ Wq[0:256]             (512 blocks x 256 thr)
// ---------------------------------------------------------------------------
__global__ void qu_kernel(const float* __restrict__ UPT, const int* __restrict__ user,
                          const float* __restrict__ Wq, float* __restrict__ QU) {
  __shared__ float u[DD];
  int b = blockIdx.x, j = threadIdx.x;
  long long uid = (long long)user[b];
  u[j] = UPT[uid * DD + j];
  __syncthreads();
  float a = 0.f;
#pragma unroll 4
  for (int d = 0; d < DD; ++d) a = fmaf(u[d], Wq[d * DD + j], a);
  QU[b * DD + j] = a;
}

// ---------------------------------------------------------------------------
// K3: pack operands into WMMA register layouts (one thread per (tile,lane)).
// tiles [0,128)   WuB: B-layout Wu (256x256) (tile = ct*8 + kt, kt<8)
// tiles [128,144) KA : A-layout K rows       (tile = h*4 + mt*2 + kt; M=t pad)
// tiles [144,160) VB : B-layout V (pad 32)   (tile = ct;  B[t,d]=v[t,d])
// B layout: lane = column N; lanes 16-31 hold K+16; 16 contiguous halves.
// A layout: lane = row M;    lanes 16-31 hold K+16; 16 contiguous halves (see load_a_rm).
// ---------------------------------------------------------------------------
__global__ void pack_kernel(const float* __restrict__ Wu, const float* __restrict__ Kf,
                            const float* __restrict__ Vf, _Float16* __restrict__ WuB,
                            _Float16* __restrict__ KA, _Float16* __restrict__ VB) {
  int g = blockIdx.x * blockDim.x + threadIdx.x;
  if (g >= 160 * 32) return;
  int lane = g & 31;
  int tIdx = g >> 5;
  int kbB  = (lane >> 4) << 4;  // B layout: 0 or 16
  int kbA  = (lane >> 4) << 3;  // unused: 0 or 8
  int coll = lane & 15;
  if (tIdx < 128) {
    int ct = tIdx >> 3, kt = tIdx & 7;
    _Float16* dst = WuB + (((tIdx << 5) + lane) << 4);
#pragma unroll
    for (int j = 0; j < 16; ++j)
      dst[j] = (_Float16)Wu[(kt * 32 + kbB + j) * DD + ct * 16 + coll];
  } else if (tIdx < 144) {
    int ti = tIdx - 128;
    int h = ti >> 2, mt = (ti >> 1) & 1, kt = ti & 1;
    _Float16* dst = KA + (((ti << 5) + lane) << 4);
    int t = mt * 16 + coll;               // A row = timeslot
#pragma unroll
    for (int j = 0; j < 16; ++j) {
      int kk = kt * 32 + kbB + j;   // contiguous K per lane half
      dst[j] = (t < TT) ? (_Float16)Kf[t * DD + h * HD + kk] : (_Float16)0.0f;
    }
  } else {
    int ti = tIdx - 144;  // d-tile 0..15
    _Float16* dst = VB + (((ti << 5) + lane) << 4);
    int col = ti * 16 + coll;
#pragma unroll
    for (int j = 0; j < 16; ++j) {
      int tk = kbB + j;
      dst[j] = (tk < TT) ? (_Float16)Vf[tk * DD + col] : (_Float16)0.0f;
    }
  }
}

// ---------------------------------------------------------------------------
// K4: fused attention. 32 rows per block, 4 waves (one head each, 2 row tiles).
// ---------------------------------------------------------------------------
__global__ __launch_bounds__(128)
void attn_kernel(const float* __restrict__ QU, const float* __restrict__ P,
                 const _Float16* __restrict__ KA, const _Float16* __restrict__ VB,
                 const _Float16* __restrict__ WuB, const int* __restrict__ hour,
                 const int* __restrict__ mask, const float* __restrict__ bu,
                 float* __restrict__ out) {
  __shared__ __align__(16) _Float16 q_lds[32 * DD];          // 16 KB
  __shared__ __align__(16) float    bias_lds[32 * TP];       //  4 KB
  __shared__ __align__(16) _Float16 attn_lds[NH * 2][16 * TP]; // 8 KB
  __shared__ __align__(16) _Float16 heads_lds[32 * DD];      // 16 KB

  int row0 = blockIdx.x << 5;
  int tid  = threadIdx.x;

  // Phase 0: q = QU[b] + P[hour[n]] -> f16 LDS; mask -> bias table (32 rows).
  {
    int r  = tid >> 2;        // 0..31
    int c4 = tid & 3;
    int n  = row0 + r;
    const float* qu = QU + (n >> 7) * DD;   // b = n / 128
    const float* pp = P + hour[n] * DD;
    int d0 = c4 * 64;
#pragma unroll 8
    for (int d = d0; d < d0 + 64; ++d)
      q_lds[r * DD + d] = (_Float16)(qu[d] + pp[d]);
    const int* mrow = mask + n * TT;
    int t0 = c4 * 8;
#pragma unroll
    for (int t = t0; t < t0 + 8; ++t)
      bias_lds[r * TP + t] = (t < TT) ? ((mrow[t] == 1) ? -1e9f : 0.0f)
                                      : -3.0e38f;
  }
  __syncthreads();

  int lane = tid & 31;
  int head = tid >> 5;
  int half = lane >> 4;
  int col  = lane & 15;
  const v8f z = {0.f, 0.f, 0.f, 0.f, 0.f, 0.f, 0.f, 0.f};
  const float scale = 0.125f;  // 1/sqrt(64)

  // Phase 1: per row-tile: scoresT -> softmax -> attn@V -> heads to LDS.
#pragma unroll 1
  for (int rt = 0; rt < 2; ++rt) {
    const _Float16* qrow = q_lds + (rt * 16 + col) * DD + head * HD + ((lane >> 4) << 4);
    // scoresT = K_h (32x64 A, T padded) x q_h^T (64x16 B).
    // C-layout: lane = query row n; regs = timeslot t.
    v8f st0 = z, st1 = z;
#pragma unroll
    for (int kt = 0; kt < 2; ++kt) {
      const _Float16* qp = qrow + kt * 32;
      v16h qb = cat8(*(const v8h*)qp, *(const v8h*)(qp + 8));      // q^T as B
      v16h a0 = load_pk(KA, head * 4 + 0 + kt, lane);              // t = 0..15
      v16h a1 = load_pk(KA, head * 4 + 2 + kt, lane);              // t = 16..31
      st0 = __builtin_amdgcn_wmma_f32_16x16x32_f16(false, a0, false, qb, (short)0, st0, false, false);
      st1 = __builtin_amdgcn_wmma_f32_16x16x32_f16(false, a1, false, qb, (short)0, st1, false, false);
    }

    // Branch-free masked softmax; one lane owns row n = col.
    // t for st0[r] = half*8 + r, st1[r] = 16 + half*8 + r; rest in lane^16.
    const float* brow = bias_lds + (rt * 16 + col) * TP;
    float sv0[8], sv1[8];
    float m = -3.0e38f;
#pragma unroll
    for (int r = 0; r < 8; ++r) {
      float b0 = brow[half * 8 + r];
      float b1 = brow[16 + half * 8 + r];
      sv0[r] = (b0 != 0.0f) ? b0 : st0[r] * scale;  // masked -> -1e9 (as ref)
      sv1[r] = (b1 != 0.0f) ? b1 : st1[r] * scale;  // pad    -> -3e38
      m = fmaxf(m, fmaxf(sv0[r], sv1[r]));
    }
    m = fmaxf(m, swz16(m));
    float e0[8], e1[8];
    float s = 0.0f;
#pragma unroll
    for (int r = 0; r < 8; ++r) {
      e0[r] = __expf(sv0[r] - m);
      e1[r] = __expf(sv1[r] - m);
      s += e0[r] + e1[r];
    }
    s += swz16(s);
    float inv = 1.0f / s;
    v8h h0, h1;
#pragma unroll
    for (int r = 0; r < 8; ++r) {
      h0[r] = (_Float16)(e0[r] * inv);
      h1[r] = (_Float16)(e1[r] * inv);
    }
    _Float16* aw = &attn_lds[head * 2 + rt][0];
    *(v8h*)(aw + col * TP + half * 8)      = h0;   // attn[n][t], vectorized
    *(v8h*)(aw + col * TP + 16 + half * 8) = h1;

    // heads_h = attn (16x32 A) @ V_h (32x64 B); stage heads in LDS.
    v16h aat = load_a_rm(aw, TP, lane);
    v8f hv[4];
#pragma unroll
    for (int ci = 0; ci < 4; ++ci) {
      v16h b = load_pk(VB, head * 4 + ci, lane);
      hv[ci] = __builtin_amdgcn_wmma_f32_16x16x32_f16(false, aat, false, b, (short)0, z, false, false);
    }
#pragma unroll
    for (int ci = 0; ci < 4; ++ci) {
      int c = head * HD + ci * 16 + col;
#pragma unroll
      for (int r = 0; r < 8; ++r)
        heads_lds[(rt * 16 + r + half * 8) * DD + c] = (_Float16)hv[ci][r];
    }
  }
  __syncthreads();

  // Phase 2: out = heads (32x256) @ Wu (256x256) + bu; each wave: 64 cols,
  // two row tiles per B load (B reuse x2).
  v8f acc0[4] = {z, z, z, z};
  v8f acc1[4] = {z, z, z, z};
#pragma unroll 2
  for (int kt = 0; kt < KT8; ++kt) {
    v16h a0 = load_a_rm(heads_lds + kt * 32, DD, lane);
    v16h a1 = load_a_rm(heads_lds + 16 * DD + kt * 32, DD, lane);
#pragma unroll
    for (int ci = 0; ci < 4; ++ci) {
      v16h b = load_pk(WuB, (head * 4 + ci) * KT8 + kt, lane);
      acc0[ci] = __builtin_amdgcn_wmma_f32_16x16x32_f16(false, a0, false, b, (short)0, acc0[ci], false, false);
      acc1[ci] = __builtin_amdgcn_wmma_f32_16x16x32_f16(false, a1, false, b, (short)0, acc1[ci], false, false);
    }
  }
#pragma unroll
  for (int ci = 0; ci < 4; ++ci) {
    int c = (head * 4 + ci) * 16 + col;
    float bval = bu[c];
#pragma unroll
    for (int r = 0; r < 8; ++r) {
      int M = r + half * 8;
      out[(long long)(row0 + M) * DD + c]      = acc0[ci][r] + bval;
      out[(long long)(row0 + 16 + M) * DD + c] = acc1[ci][r] + bval;
    }
  }
}

// ---------------------------------------------------------------------------
extern "C" void kernel_launch(void* const* d_in, const int* in_sizes, int n_in,
                              void* d_out, int out_size, void* d_ws, size_t ws_size,
                              hipStream_t stream) {
  const float* E    = (const float*)d_in[0];
  const int*   user = (const int*)  d_in[1];
  const int*   hour = (const int*)  d_in[2];
  const int*   mask = (const int*)  d_in[3];
  const float* UPT  = (const float*)d_in[4];
  const float* Wq   = (const float*)d_in[5];
  const float* bq   = (const float*)d_in[6];
  const float* Wk   = (const float*)d_in[7];
  const float* bk   = (const float*)d_in[8];
  const float* Wv   = (const float*)d_in[9];
  const float* bv   = (const float*)d_in[10];
  const float* Wu   = (const float*)d_in[11];
  const float* bu   = (const float*)d_in[12];
  float* out = (float*)d_out;
  (void)in_sizes; (void)n_in; (void)out_size; (void)ws_size;

  char* ws = (char*)d_ws;
  size_t off = 0;
  auto alloc = [&](size_t bytes) -> void* {
    void* p = ws + off;
    off = (off + bytes + 255) & ~(size_t)255;
    return p;
  };
  float*    P   = (float*)   alloc(TT * DD * sizeof(float));
  float*    Kf  = (float*)   alloc(TT * DD * sizeof(float));
  float*    Vf  = (float*)   alloc(TT * DD * sizeof(float));
  float*    QU  = (float*)   alloc(512 * DD * sizeof(float));
  _Float16* WuB = (_Float16*)alloc(128 * 32 * 16 * sizeof(_Float16));
  _Float16* KA  = (_Float16*)alloc(16 * 32 * 16 * sizeof(_Float16));
  _Float16* VB  = (_Float16*)alloc(16 * 32 * 16 * sizeof(_Float16));

  hipLaunchKernelGGL(kv_p_kernel, dim3(TT), dim3(DD), 0, stream,
                     E, Wq, bq, Wk, bk, Wv, bv, Kf, Vf, P);
  hipLaunchKernelGGL(qu_kernel, dim3(512), dim3(DD), 0, stream, UPT, user, Wq, QU);
  hipLaunchKernelGGL(pack_kernel, dim3(40), dim3(128), 0, stream, Wu, Kf, Vf, WuB, KA, VB);
  hipLaunchKernelGGL(attn_kernel, dim3(NN / 32), dim3(128), 0, stream,
                     QU, P, KA, VB, WuB, hour, mask, bu, out);
}
